// ResidualAttentionLayer_37709812859059
// MI455X (gfx1250) — matrix operand order repaired and checked by reading, not compiled, not run on hardware
//
#include <hip/hip_runtime.h>
#include <hip/hip_bf16.h>
#include <cstdint>
#include <cstddef>

// ---------------------------------------------------------------------------
// MI455X (gfx1250) residual attention layer.
// b=2, t=2048, D=1024, H=16, hd=64, FF=4096.  All matmuls via
// v_wmma_f32_16x16x32_f16 (wave32).  d_out = [x (4096*1024 f32) | attn
// (2*16*2048*2048 f32)].  The attn region of d_out doubles as logits scratch
// (raw logits written in pass 1, overwritten with softmax probs in pass 2),
// so every byte of d_out we read is written first each launch (replay-safe).
// ---------------------------------------------------------------------------

typedef __attribute__((ext_vector_type(16))) _Float16 v16h;
typedef __attribute__((ext_vector_type(8)))  float    v8f;

union FragA { v16h v; uint4 q[2]; };

__device__ __forceinline__ v8f wmma16(const FragA& a, const FragA& b, v8f c) {
  return __builtin_amdgcn_wmma_f32_16x16x32_f16(false, a.v, false, b.v,
                                                (short)0, c, false, false);
}

static __device__ __forceinline__ v8f vzero8() {
  v8f z = {0.f, 0.f, 0.f, 0.f, 0.f, 0.f, 0.f, 0.f};
  return z;
}

// ---------------------------------------------------------------------------
// f32 -> f16 elementwise convert (weights)
// ---------------------------------------------------------------------------
__global__ __launch_bounds__(256) void cvt_f32_to_f16(const float* __restrict__ in,
                                                      _Float16* __restrict__ out,
                                                      int n) {
  int i = blockIdx.x * 256 + threadIdx.x;
  if (i < n) out[i] = (_Float16)in[i];
}

// ---------------------------------------------------------------------------
// LayerNorm over D=1024, one block (8 waves) per row, output f16
// ---------------------------------------------------------------------------
__global__ __launch_bounds__(256) void ln_to_f16(const float* __restrict__ x,
                                                 const float* __restrict__ g,
                                                 const float* __restrict__ b,
                                                 _Float16* __restrict__ out) {
  int row = blockIdx.x;
  int t = threadIdx.x;
  const float* xr = x + (size_t)row * 1024;
  float v0[4];
  float s = 0.f, ss = 0.f;
#pragma unroll
  for (int i = 0; i < 4; i++) {
    float v = xr[t + 256 * i];
    v0[i] = v; s += v; ss += v * v;
  }
#pragma unroll
  for (int m = 16; m >= 1; m >>= 1) {
    s  += __shfl_xor(s,  m, 32);
    ss += __shfl_xor(ss, m, 32);
  }
  __shared__ float red[16];
  int wid = t >> 5, lid = t & 31;
  if (lid == 0) { red[wid] = s; red[8 + wid] = ss; }
  __syncthreads();
  s = 0.f; ss = 0.f;
#pragma unroll
  for (int w = 0; w < 8; w++) { s += red[w]; ss += red[8 + w]; }
  float mean = s * (1.f / 1024.f);
  float var  = ss * (1.f / 1024.f) - mean * mean;
  float inv  = rsqrtf(var + 1e-5f);
#pragma unroll
  for (int i = 0; i < 4; i++) {
    int c = t + 256 * i;
    out[(size_t)row * 1024 + c] = (_Float16)((v0[i] - mean) * inv * g[c] + b[c]);
  }
}

// ---------------------------------------------------------------------------
// Generic f16 WMMA GEMM: C[M,N] = A[M,K] (f16, row-major) * B[K,N] (f16,
// row-major).  Block tile 128x128, BK=32, 8 waves (2x4), wave tile 64x32.
// MODE 0: scatter C (f16) into q/k/v head-major buffers
// MODE 1: outf = resid + C                       (f32)
// MODE 2: outh = gelu(C + bias)                  (f16, ldout = 4096)
// MODE 3: outf = resid + C + bias                (f32)
// ---------------------------------------------------------------------------
#define LDST 40  // padded LDS row stride (f16 elems) -> conflict-free frag reads

template <int MODE>
__global__ __launch_bounds__(256) void gemm_f16(
    const _Float16* __restrict__ A, const _Float16* __restrict__ B,
    int M, int N, int K,
    const float* __restrict__ bias, const float* __restrict__ resid,
    float* __restrict__ outf, _Float16* __restrict__ outh,
    _Float16* __restrict__ qf, _Float16* __restrict__ kf,
    _Float16* __restrict__ vf) {
  __shared__ __align__(16) _Float16 lds[128 * LDST * 2];
  _Float16* As = lds;               // [128 m][LDST] (k contiguous)
  _Float16* Bt = lds + 128 * LDST;  // [128 n][LDST] (k contiguous, transposed)

  int t = threadIdx.x;
  int l = t & 31, w = t >> 5;
  int waveM = w >> 2;   // 0..1
  int waveN = w & 3;    // 0..3
  int half = l >> 4, lc = l & 15;
  int bm = blockIdx.y * 128;
  int bn = blockIdx.x * 128;

  v8f acc[4][2];
#pragma unroll
  for (int mt = 0; mt < 4; mt++)
#pragma unroll
    for (int nt = 0; nt < 2; nt++) acc[mt][nt] = vzero8();

  int arow = t >> 1, akoff = (t & 1) * 16;   // A stager: 128 rows x 32 k
  int brow = t >> 3, bnoff = (t & 7) * 16;   // B stager: 32 k-rows x 128 n

  for (int k0 = 0; k0 < K; k0 += 32) {
    // stage A tile
    const _Float16* ga = A + (size_t)(bm + arow) * K + k0 + akoff;
    uint4 a0 = *(const uint4*)ga;
    uint4 a1 = *(const uint4*)(ga + 8);
    *(uint4*)(As + arow * LDST + akoff)     = a0;
    *(uint4*)(As + arow * LDST + akoff + 8) = a1;
    // stage B tile transposed
    const _Float16* gb = B + (size_t)(k0 + brow) * N + bn + bnoff;
    uint4 b0 = *(const uint4*)gb;
    uint4 b1 = *(const uint4*)(gb + 8);
    __align__(16) _Float16 tmp[16];
    *(uint4*)tmp = b0; *(uint4*)(tmp + 8) = b1;
#pragma unroll
    for (int i = 0; i < 16; i++) Bt[(bnoff + i) * LDST + brow] = tmp[i];
    if (k0 + 32 < K) {
      __builtin_prefetch(A + (size_t)(bm + arow) * K + k0 + 32 + akoff, 0, 1);
      __builtin_prefetch(B + (size_t)(k0 + 32 + brow) * N + bn + bnoff, 0, 1);
    }
    __syncthreads();

    FragA af[4], bfr[2];
#pragma unroll
    for (int mt = 0; mt < 4; mt++) {
      const _Float16* p = As + (waveM * 64 + mt * 16 + lc) * LDST + half * 8;
      af[mt].q[0] = *(const uint4*)p;         // k = half*8 + 0..7
      af[mt].q[1] = *(const uint4*)(p + 16);  // k = 16 + half*8 + 0..7
    }
#pragma unroll
    for (int nt = 0; nt < 2; nt++) {
      const _Float16* p = Bt + (waveN * 32 + nt * 16 + lc) * LDST + half * 8;
      bfr[nt].q[0] = *(const uint4*)p;        // k = half*8 + 0..7
      bfr[nt].q[1] = *(const uint4*)(p + 16); // k = 16 + half*8 + 0..7
    }
#pragma unroll
    for (int mt = 0; mt < 4; mt++)
#pragma unroll
      for (int nt = 0; nt < 2; nt++)
        acc[mt][nt] = wmma16(af[mt], bfr[nt], acc[mt][nt]);
    __syncthreads();
  }

  // epilogue (D layout: reg r -> m = r + 8*half, lane -> n = lc)
#pragma unroll
  for (int mt = 0; mt < 4; mt++) {
#pragma unroll
    for (int nt = 0; nt < 2; nt++) {
      int gmBase = bm + waveM * 64 + mt * 16 + 8 * half;
      int gn = bn + waveN * 32 + nt * 16 + lc;
#pragma unroll
      for (int r = 0; r < 8; r++) {
        int gm = gmBase + r;
        float v = acc[mt][nt][r];
        if (MODE == 0) {
          int mat = gn >> 10, rem = gn & 1023;
          int head = rem >> 6, d = rem & 63;
          int bidx = gm >> 11, tloc = gm & 2047;
          size_t idx = ((size_t)(bidx * 16 + head) * 2048 + tloc) * 64 + d;
          _Float16 hv = (_Float16)v;
          (mat == 0 ? qf : (mat == 1 ? kf : vf))[idx] = hv;
        } else if (MODE == 1) {
          size_t idx = (size_t)gm * 1024 + gn;
          outf[idx] = resid[idx] + v;
        } else if (MODE == 2) {
          float z = v + bias[gn];
          float ge = 0.5f * z * (1.0f + erff(z * 0.70710678f));
          outh[(size_t)gm * 4096 + gn] = (_Float16)ge;
        } else {
          size_t idx = (size_t)gm * 1024 + gn;
          outf[idx] = resid[idx] + v + bias[gn];
        }
      }
    }
  }
}

// ---------------------------------------------------------------------------
// Fused attention: per block = one (b,h) x 128 query rows; per wave = 16 rows.
// Pass 1: S = q K^T * scale + prev  -> raw logits to d_out, online max/sum.
// Pass 2: re-read logits, write softmax probs to d_out, accumulate P*V.
// ---------------------------------------------------------------------------
__global__ __launch_bounds__(256) void attention_fused(
    const _Float16* __restrict__ qf, const _Float16* __restrict__ kf,
    const _Float16* __restrict__ vf, const float* __restrict__ prev,
    float* __restrict__ attn, _Float16* __restrict__ attnout) {
  __shared__ __align__(16) _Float16 smem[7680];  // phase1: K tile 64x72
                                                 // phase2: Vt 64x40 + P 8x16x40
  int t = threadIdx.x;
  int l = t & 31, w = t >> 5;
  int half = l >> 4, lc = l & 15;
  int bh = blockIdx.y;                      // b*16 + h
  int mBase = blockIdx.x * 128 + w * 16;    // query row base (t index)
  size_t bhQ = (size_t)bh * 2048 * 64;
  size_t bhA = (size_t)bh * 2048 * 2048;
  const float scale = 0.125f;               // 64^-0.5

  // Q fragments for rows mBase..mBase+15, k-chunks d in [0,32) and [32,64)
  FragA aq0, aq1;
  {
    const _Float16* p = qf + bhQ + (size_t)(mBase + lc) * 64;
    aq0.q[0] = *(const uint4*)(p + half * 8);
    aq0.q[1] = *(const uint4*)(p + 16 + half * 8);
    aq1.q[0] = *(const uint4*)(p + 32 + half * 8);
    aq1.q[1] = *(const uint4*)(p + 48 + half * 8);
  }

  float rmax[8], rsum[8];
#pragma unroll
  for (int r = 0; r < 8; r++) { rmax[r] = -3.0e38f; rsum[r] = 0.f; }

  // ---- pass 1: logits + online softmax stats ----
  _Float16* ldsK = smem;  // [64 n][72] (d contiguous)
  for (int n0 = 0; n0 < 2048; n0 += 64) {
    {
      int krow = t >> 2, koff = (t & 3) * 16;
      const _Float16* g = kf + bhQ + (size_t)(n0 + krow) * 64 + koff;
      uint4 u0 = *(const uint4*)g;
      uint4 u1 = *(const uint4*)(g + 8);
      *(uint4*)(ldsK + krow * 72 + koff)     = u0;
      *(uint4*)(ldsK + krow * 72 + koff + 8) = u1;
    }
    __syncthreads();
#pragma unroll
    for (int nt = 0; nt < 4; nt++) {
      FragA bk0, bk1;
      const _Float16* p = ldsK + (nt * 16 + lc) * 72 + half * 8;
      bk0.q[0] = *(const uint4*)p;          // d = half*8 + 0..7
      bk0.q[1] = *(const uint4*)(p + 16);   // d = 16 + half*8 + 0..7
      bk1.q[0] = *(const uint4*)(p + 32);   // d = 32 + half*8 + 0..7
      bk1.q[1] = *(const uint4*)(p + 48);
      v8f s = vzero8();
      s = wmma16(aq0, bk0, s);
      s = wmma16(aq1, bk1, s);

      size_t rowbase = bhA + (size_t)(mBase + 8 * half) * 2048 + (n0 + nt * 16 + lc);
      float vv[8];
#pragma unroll
      for (int r = 0; r < 8; r++) {
        float v = s[r] * scale + prev[rowbase + (size_t)r * 2048];
        attn[rowbase + (size_t)r * 2048] = v;  // raw logits (scratch)
        vv[r] = v;
      }
#pragma unroll
      for (int r = 0; r < 8; r++) {
        float m = vv[r];
#pragma unroll
        for (int msk = 8; msk >= 1; msk >>= 1) m = fmaxf(m, __shfl_xor(m, msk, 16));
        float nm = fmaxf(rmax[r], m);
        float e = __expf(vv[r] - nm);
#pragma unroll
        for (int msk = 8; msk >= 1; msk >>= 1) e += __shfl_xor(e, msk, 16);
        rsum[r] = rsum[r] * __expf(rmax[r] - nm) + e;
        rmax[r] = nm;
      }
    }
    __syncthreads();
  }

  float rinv[8];
#pragma unroll
  for (int r = 0; r < 8; r++) rinv[r] = 1.0f / rsum[r];

  // ---- pass 2: normalize + P*V ----
  _Float16* ldsV = smem;                           // [64 d][40] (n contiguous)
  _Float16* ldsP = smem + 64 * 40 + w * 16 * 40;   // per-wave [16 m][40]
  v8f o[4];
#pragma unroll
  for (int dt = 0; dt < 4; dt++) o[dt] = vzero8();

  for (int n0 = 0; n0 < 2048; n0 += 32) {
    __syncthreads();
    {  // stage V transposed: Vt[d][n]
      int vn = t >> 3, vdo = (t & 7) * 8;
      const _Float16* g = vf + bhQ + (size_t)(n0 + vn) * 64 + vdo;
      __align__(16) _Float16 tmp[8];
      *(uint4*)tmp = *(const uint4*)g;
#pragma unroll
      for (int i = 0; i < 8; i++) ldsV[(vdo + i) * 40 + vn] = tmp[i];
    }
    // normalize logits, emit final attn, pack P into LDS (A layout source)
#pragma unroll
    for (int nn = 0; nn < 2; nn++) {
      size_t rowbase = bhA + (size_t)(mBase + 8 * half) * 2048 + (n0 + nn * 16 + lc);
#pragma unroll
      for (int r = 0; r < 8; r++) {
        float v = attn[rowbase + (size_t)r * 2048];
        float p = __expf(v - rmax[r]) * rinv[r];
        attn[rowbase + (size_t)r * 2048] = p;              // final attn output
        ldsP[(r + 8 * half) * 40 + nn * 16 + lc] = (_Float16)p;
      }
    }
    __syncthreads();

    FragA ap;
    const _Float16* pp = ldsP + lc * 40 + half * 8;
    ap.q[0] = *(const uint4*)pp;
    ap.q[1] = *(const uint4*)(pp + 16);
#pragma unroll
    for (int dt = 0; dt < 4; dt++) {
      FragA bv;
      const _Float16* p = ldsV + (dt * 16 + lc) * 40 + half * 8;
      bv.q[0] = *(const uint4*)p;
      bv.q[1] = *(const uint4*)(p + 16);
      o[dt] = wmma16(ap, bv, o[dt]);
    }
  }

  // write attn-out rows head-major: [b*2048 + t][h*64 + d] (f16)
  int bidx = bh >> 4, head = bh & 15;
#pragma unroll
  for (int dt = 0; dt < 4; dt++) {
#pragma unroll
    for (int r = 0; r < 8; r++) {
      int m = mBase + r + 8 * half;
      int d = dt * 16 + lc;
      attnout[(size_t)(bidx * 2048 + m) * 1024 + head * 64 + d] = (_Float16)o[dt][r];
    }
  }
}

// ---------------------------------------------------------------------------
// host side
// ---------------------------------------------------------------------------
extern "C" void kernel_launch(void* const* d_in, const int* in_sizes, int n_in,
                              void* d_out, int out_size, void* d_ws, size_t ws_size,
                              hipStream_t stream) {
  const float* x      = (const float*)d_in[0];
  const float* prev   = (const float*)d_in[1];
  const float* w_qkv  = (const float*)d_in[2];
  const float* w_out  = (const float*)d_in[3];
  const float* w1     = (const float*)d_in[4];
  const float* b1     = (const float*)d_in[5];
  const float* w2     = (const float*)d_in[6];
  const float* b2     = (const float*)d_in[7];
  const float* g1     = (const float*)d_in[8];
  const float* be1    = (const float*)d_in[9];
  const float* g2     = (const float*)d_in[10];
  const float* be2    = (const float*)d_in[11];

  float* xout = (float*)d_out;              // 4096*1024
  float* attn = xout + (size_t)4096 * 1024; // 2*16*2048*2048

  // workspace carve-up (f16 unless noted); total ~126 MB
  char* p = (char*)d_ws;
  auto take = [&](size_t elems, size_t esz) { char* q = p; p += elems * esz; return q; };
  _Float16* wqkv_h = (_Float16*)take((size_t)1024 * 3072, 2);
  _Float16* wout_h = (_Float16*)take((size_t)1024 * 1024, 2);
  _Float16* w1_h   = (_Float16*)take((size_t)1024 * 4096, 2);
  _Float16* w2_h   = (_Float16*)take((size_t)4096 * 1024, 2);
  _Float16* ln1_h  = (_Float16*)take((size_t)4096 * 1024, 2);
  _Float16* q_h    = (_Float16*)take((size_t)4096 * 1024, 2);
  _Float16* k_h    = (_Float16*)take((size_t)4096 * 1024, 2);
  _Float16* v_h    = (_Float16*)take((size_t)4096 * 1024, 2);
  _Float16* ao_h   = (_Float16*)take((size_t)4096 * 1024, 2);
  _Float16* ln2_h  = (_Float16*)take((size_t)4096 * 1024, 2);
  _Float16* ffh_h  = (_Float16*)take((size_t)4096 * 4096, 2);
  float*    x2_f   = (float*)   take((size_t)4096 * 1024, 4);

  // 1. weight conversion
  {
    int n;
    n = 1024 * 3072; cvt_f32_to_f16<<<(n + 255) / 256, 256, 0, stream>>>(w_qkv, wqkv_h, n);
    n = 1024 * 1024; cvt_f32_to_f16<<<(n + 255) / 256, 256, 0, stream>>>(w_out, wout_h, n);
    n = 1024 * 4096; cvt_f32_to_f16<<<(n + 255) / 256, 256, 0, stream>>>(w1, w1_h, n);
    n = 4096 * 1024; cvt_f32_to_f16<<<(n + 255) / 256, 256, 0, stream>>>(w2, w2_h, n);
  }
  // 2. LN1
  ln_to_f16<<<4096, 256, 0, stream>>>(x, g1, be1, ln1_h);
  // 3. QKV projection  (M=4096, N=3072, K=1024) -> head-major q/k/v
  gemm_f16<0><<<dim3(24, 32), 256, 0, stream>>>(ln1_h, wqkv_h, 4096, 3072, 1024,
                                                nullptr, nullptr, nullptr, nullptr,
                                                q_h, k_h, v_h);
  // 4. fused attention (softmax output -> d_out attn region, PV -> ao_h)
  attention_fused<<<dim3(16, 32), 256, 0, stream>>>(q_h, k_h, v_h, prev, attn, ao_h);
  // 5. out projection + residual: x2 = x + ao @ w_out
  gemm_f16<1><<<dim3(8, 32), 256, 0, stream>>>(ao_h, wout_h, 4096, 1024, 1024,
                                               nullptr, x, x2_f, nullptr,
                                               nullptr, nullptr, nullptr);
  // 6. LN2
  ln_to_f16<<<4096, 256, 0, stream>>>(x2_f, g2, be2, ln2_h);
  // 7. FFN1 + GELU
  gemm_f16<2><<<dim3(32, 32), 256, 0, stream>>>(ln2_h, w1_h, 4096, 4096, 1024,
                                                b1, nullptr, nullptr, ffh_h,
                                                nullptr, nullptr, nullptr);
  // 8. FFN2 + bias + residual -> final x
  gemm_f16<3><<<dim3(8, 32), 256, 0, stream>>>(ffh_h, w2_h, 4096, 1024, 4096,
                                               b2, x2_f, xout, nullptr,
                                               nullptr, nullptr, nullptr);
}
